// DeformConv_60129542838
// MI455X (gfx1250) — compile-verified, matrix operand order checked by reading, not run on hardware
//
#include <hip/hip_runtime.h>
#include <hip/hip_bf16.h>

// ---------------------------------------------------------------------------
// Problem constants (from reference)
// ---------------------------------------------------------------------------
#define BATCH 2
#define CI    256
#define CO    256
#define Hd    128
#define Wd    128
#define HW    (Hd * Wd)          // 16384
#define NPX   (BATCH * HW)       // 32768 pixels total
#define NK    9                  // 3x3 taps
#define BN_EPS 1e-5f

// GEMM tiling for the WMMA kernel
#define CO_TILE 128              // per workgroup
#define PX_TILE 32               // per workgroup
#define KCHUNK  32               // bf16 WMMA K

typedef __attribute__((ext_vector_type(16))) __bf16        v16bf;
typedef __attribute__((ext_vector_type(8)))  float         v8f;
typedef __attribute__((ext_vector_type(8)))  unsigned int  v8u;

// float -> bf16 bits, round-to-nearest-even
__device__ __forceinline__ unsigned short f2bf(float f) {
    unsigned int u = __builtin_bit_cast(unsigned int, f);
    unsigned int r = u + 0x7FFFu + ((u >> 16) & 1u);
    return (unsigned short)(r >> 16);
}

// load a 16-element bf16 fragment (32 bytes, caller guarantees 32B alignment)
__device__ __forceinline__ v16bf load_frag(const unsigned short* p) {
    return __builtin_bit_cast(v16bf, *reinterpret_cast<const v8u*>(p));
}

// ---------------------------------------------------------------------------
// Kernel 1: w_dcn [CO][CI][3][3] f32  ->  wk [9][CO][CI] bf16 (contiguous K)
// ---------------------------------------------------------------------------
__global__ __launch_bounds__(256)
void prep_weights_kernel(const float* __restrict__ w_dcn,
                         unsigned short* __restrict__ wk) {
    const int idx = blockIdx.x * 256 + threadIdx.x;   // [k][co][ci]
    if (idx >= NK * CO * CI) return;
    const int ci = idx & (CI - 1);
    const int co = (idx >> 8) & (CO - 1);
    const int k  = idx >> 16;
    wk[idx] = f2bf(w_dcn[(co * CI + ci) * NK + k]);
}

// ---------------------------------------------------------------------------
// Kernel 2: offset conv (27 ch, 3x3, pad 1) -> dy/dx/sigmoid(mask) planes
// One thread per output pixel; w_off staged through LDS in 32-ci chunks.
// ---------------------------------------------------------------------------
__global__ __launch_bounds__(256)
void offset_conv_kernel(const float* __restrict__ x,
                        const float* __restrict__ w_off,
                        const float* __restrict__ b_off,
                        float* __restrict__ g_dy,
                        float* __restrict__ g_dx,
                        float* __restrict__ g_mask) {
    __shared__ float s_w[27 * 32 * NK];               // 31104 B
    const int tid = threadIdx.x;
    const int p   = blockIdx.x * 256 + tid;
    const int b   = p >> 14;
    const int hw  = p & (HW - 1);
    const int h   = hw >> 7;
    const int w   = hw & (Wd - 1);

    float acc[27];
#pragma unroll
    for (int c = 0; c < 27; ++c) acc[c] = 0.f;

    for (int cc = 0; cc < CI / 32; ++cc) {
        __syncthreads();
        // stage w_off chunk: s_w[co*288 + ci_l*9 + t]
        for (int idx = tid; idx < 27 * 32 * NK; idx += 256) {
            const int co  = idx / 288;
            const int rem = idx - co * 288;
            s_w[idx] = w_off[co * (CI * NK) + cc * 288 + rem];
        }
        __syncthreads();

        const float* xb = x + (size_t)(b * CI + cc * 32) * HW;
        for (int cl = 0; cl < 32; ++cl) {
            const float* xc = xb + (size_t)cl * HW;
            float xv[NK];
#pragma unroll
            for (int t = 0; t < NK; ++t) {
                const int yy = h + t / 3 - 1;
                const int xx = w + t % 3 - 1;
                const bool ok = (yy >= 0) & (yy < Hd) & (xx >= 0) & (xx < Wd);
                xv[t] = ok ? xc[yy * Wd + xx] : 0.f;
            }
#pragma unroll
            for (int co = 0; co < 27; ++co) {
                const float* wp = &s_w[co * 288 + cl * NK];
#pragma unroll
                for (int t = 0; t < NK; ++t)
                    acc[co] = fmaf(wp[t], xv[t], acc[co]);
            }
        }
    }

#pragma unroll
    for (int k = 0; k < NK; ++k) {
        const float dyv = acc[2 * k]     + b_off[2 * k];
        const float dxv = acc[2 * k + 1] + b_off[2 * k + 1];
        float mm        = acc[18 + k]    + b_off[18 + k];
        mm = 1.f / (1.f + __expf(-mm));
        g_dy[k * NPX + p]   = dyv;
        g_dx[k * NPX + p]   = dxv;
        g_mask[k * NPX + p] = mm;
    }
}

// ---------------------------------------------------------------------------
// Kernel 3: deformable gather + 9 GEMMs via v_wmma_f32_16x16x32_bf16
//           + fused bias/BN/ReLU epilogue.
// 256 threads (8 waves). Tile: 128 CO x 32 px. LDS holds bf16 val[32px][256ci].
// ---------------------------------------------------------------------------
__global__ __launch_bounds__(256)
void dcn_wmma_kernel(const float* __restrict__ x,
                     const unsigned short* __restrict__ wk,   // [9][CO][CI] bf16
                     const float* __restrict__ g_dy,
                     const float* __restrict__ g_dx,
                     const float* __restrict__ g_mask,
                     const float* __restrict__ b_dcn,
                     const float* __restrict__ gamma,
                     const float* __restrict__ beta,
                     const float* __restrict__ run_mean,
                     const float* __restrict__ run_var,
                     float* __restrict__ out) {
    __shared__ alignas(32) unsigned short sVal[PX_TILE * CI];  // 16 KB

    const int tid  = threadIdx.x;
    const int lane = tid & 31;
    const int wv   = tid >> 5;

    const int co_blk = blockIdx.x & 1;          // 2 CO blocks
    const int px_blk = blockIdx.x >> 1;         // 1024 px blocks
    const int p0     = px_blk * PX_TILE;
    const int co0    = co_blk * CO_TILE + wv * 16;

    // sampling-stage role: lane-in-32 -> pixel, group -> ci range
    const int px_l = tid & 31;
    const int ci0  = (tid >> 5) * 32;
    const int p    = p0 + px_l;
    const int b    = p >> 14;
    const int hw   = p & (HW - 1);
    const int h    = hw >> 7;
    const int w    = hw & (Wd - 1);

    // WMMA fragment addressing (this code's K order for both A and B: lanes 0-15 K=0..15,
    // lanes 16-31 K=16..31; 16 contiguous K elements per lane)
    const int a_row  = co0 + (lane & 15);
    const int f_koff = (lane >> 4) * 16;
    const int b_col  = lane & 15;

    v8f c0 = {};
    v8f c1 = {};

    for (int k = 0; k < NK; ++k) {
        const int ki = k / 3, kj = k % 3;
        // ---- stage 1: bilinear sample * mask -> LDS (bf16) ----
        {
            const float dyv = g_dy[k * NPX + p];
            const float dxv = g_dx[k * NPX + p];
            const float mv  = g_mask[k * NPX + p];
            const float py  = (float)(h - 1 + ki) + dyv;
            const float pxf = (float)(w - 1 + kj) + dxv;
            const float y0f = floorf(py), x0f = floorf(pxf);
            const float ly  = py - y0f,  lx  = pxf - x0f;
            const int y0 = (int)y0f, x0 = (int)x0f;
            const int y1 = y0 + 1,  x1 = x0 + 1;
            const float vy0 = (y0 >= 0 && y0 < Hd) ? 1.f : 0.f;
            const float vy1 = (y1 >= 0 && y1 < Hd) ? 1.f : 0.f;
            const float vx0 = (x0 >= 0 && x0 < Wd) ? 1.f : 0.f;
            const float vx1 = (x1 >= 0 && x1 < Wd) ? 1.f : 0.f;
            const int y0c = y0 < 0 ? 0 : (y0 > Hd - 1 ? Hd - 1 : y0);
            const int y1c = y1 < 0 ? 0 : (y1 > Hd - 1 ? Hd - 1 : y1);
            const int x0c = x0 < 0 ? 0 : (x0 > Wd - 1 ? Wd - 1 : x0);
            const int x1c = x1 < 0 ? 0 : (x1 > Wd - 1 ? Wd - 1 : x1);
            const float w00 = (1.f - ly) * (1.f - lx) * vy0 * vx0 * mv;
            const float w01 = (1.f - ly) * lx         * vy0 * vx1 * mv;
            const float w10 = ly         * (1.f - lx) * vy1 * vx0 * mv;
            const float w11 = ly         * lx         * vy1 * vx1 * mv;
            const int i00 = y0c * Wd + x0c, i01 = y0c * Wd + x1c;
            const int i10 = y1c * Wd + x0c, i11 = y1c * Wd + x1c;

            const float* xb = x + (size_t)(b * CI + ci0) * HW;
            unsigned short* dst = &sVal[px_l * CI + ci0];
#pragma unroll 4
            for (int c = 0; c < 32; ++c) {
                const float* xc = xb + (size_t)c * HW;
                float v = w00 * xc[i00] + w01 * xc[i01] +
                          w10 * xc[i10] + w11 * xc[i11];
                dst[c] = f2bf(v);
            }
        }
        __syncthreads();

        // ---- stage 2: 8 K-chunks x 2 px-subtiles of WMMA ----
        const unsigned short* wkk = wk + (size_t)k * CO * CI;
#pragma unroll
        for (int kk = 0; kk < CI / KCHUNK; ++kk) {
            const int kb = kk * KCHUNK + f_koff;
            v16bf A  = load_frag(wkk + a_row * CI + kb);
            v16bf B0 = load_frag(&sVal[b_col * CI + kb]);
            v16bf B1 = load_frag(&sVal[(16 + b_col) * CI + kb]);
            c0 = __builtin_amdgcn_wmma_f32_16x16x32_bf16(
                     false, A, false, B0, (short)0, c0, false, false);
            c1 = __builtin_amdgcn_wmma_f32_16x16x32_bf16(
                     false, A, false, B1, (short)0, c1, false, false);
        }
        __syncthreads();
    }

    // ---- epilogue: +bias, BN (scale/shift), ReLU, scatter f32 ----
    // D layout: VGPR r -> M = r (lanes 0-15) / r+8 (lanes 16-31); N = lane%16
    const int n0 = lane & 15;
    const int mo = (lane >> 4) * 8;
#pragma unroll
    for (int r = 0; r < 8; ++r) {
        const int co  = co0 + r + mo;
        const float inv = gamma[co] * rsqrtf(run_var[co] + BN_EPS);
        const float sh  = beta[co] - run_mean[co] * inv + b_dcn[co] * inv;
        {
            const int pp  = p0 + n0;
            const int bb  = pp >> 14;
            const int rem = pp & (HW - 1);
            float v = fmaxf(c0[r] * inv + sh, 0.f);
            out[(size_t)(bb * CO + co) * HW + rem] = v;
        }
        {
            const int pp  = p0 + 16 + n0;
            const int bb  = pp >> 14;
            const int rem = pp & (HW - 1);
            float v = fmaxf(c1[r] * inv + sh, 0.f);
            out[(size_t)(bb * CO + co) * HW + rem] = v;
        }
    }
}

// ---------------------------------------------------------------------------
// Launch
// ---------------------------------------------------------------------------
extern "C" void kernel_launch(void* const* d_in, const int* in_sizes, int n_in,
                              void* d_out, int out_size, void* d_ws, size_t ws_size,
                              hipStream_t stream) {
    const float* x        = (const float*)d_in[0];
    const float* w_off    = (const float*)d_in[1];
    const float* b_off    = (const float*)d_in[2];
    const float* w_dcn    = (const float*)d_in[3];
    const float* b_dcn    = (const float*)d_in[4];
    const float* gamma    = (const float*)d_in[5];
    const float* beta     = (const float*)d_in[6];
    const float* run_mean = (const float*)d_in[7];
    const float* run_var  = (const float*)d_in[8];
    float* out = (float*)d_out;

    // workspace layout
    float* g_dy   = (float*)d_ws;                     // 9*32768 f32
    float* g_dx   = g_dy + NK * NPX;
    float* g_mask = g_dx + NK * NPX;
    unsigned short* g_wk = (unsigned short*)(g_mask + NK * NPX); // 9*256*256 bf16

    // 1) weight reorder/convert
    prep_weights_kernel<<<(NK * CO * CI) / 256, 256, 0, stream>>>(w_dcn, g_wk);

    // 2) offset conv -> dy/dx/mask planes
    offset_conv_kernel<<<NPX / 256, 256, 0, stream>>>(
        x, w_off, b_off, g_dy, g_dx, g_mask);

    // 3) gather + 9x WMMA GEMM + fused bias/BN/ReLU
    const int n_blocks = (CO / CO_TILE) * (NPX / PX_TILE);   // 2 * 1024
    dcn_wmma_kernel<<<n_blocks, 256, 0, stream>>>(
        x, g_wk, g_dy, g_dx, g_mask,
        b_dcn, gamma, beta, run_mean, run_var, out);
}
